// Transformer_performer_42219528519816
// MI455X (gfx1250) — matrix operand order repaired and checked by reading, not compiled, not run on hardware
//
#include <hip/hip_runtime.h>
#include <hip/hip_bf16.h>
#include <math.h>

// ---------------------------------------------------------------------------
// Performer attention forward for MI455X (gfx1250), wave32 + WMMA bf16.
// All heavy math runs through v_wmma_f32_16x16x32_bf16 (f32 accumulate).
// LDS double-buffered staging + global_prefetch_b8 two tiles ahead.
// ---------------------------------------------------------------------------

typedef __attribute__((ext_vector_type(16))) __bf16 v16bf;
typedef __attribute__((ext_vector_type(8)))  float  v8f;

#define BATCH 8
#define TSEQ  4096
#define DIMK  512
#define EMBD  512
#define NKQV  1536             // 3 * EMB
#define MF    256              // random-feature dim
#define MROWS (BATCH * TSEQ)   // 32768 token rows

union FragBF { v16bf v; uint4 u[2]; };
union Chunk8 { uint4 u; __bf16 e[8]; };

// ---------------------------------------------------------------------------
// small helpers
// ---------------------------------------------------------------------------
__global__ void k_cvt_f32_bf16(const float* __restrict__ in, __bf16* __restrict__ out, int n) {
  int stride = gridDim.x * blockDim.x;
  for (int i = blockIdx.x * blockDim.x + threadIdx.x; i < n; i += stride)
    out[i] = (__bf16)in[i];
}

__global__ void k_zero_f32(float* __restrict__ p, int n) {
  int stride = gridDim.x * blockDim.x;
  for (int i = blockIdx.x * blockDim.x + threadIdx.x; i < n; i += stride)
    p[i] = 0.0f;
}

// rn[row] = 0.5 * sum_d z[row,d]^2   (one wave32 per row)
__global__ void k_rownorm(const __bf16* __restrict__ z, int lda, float* __restrict__ rn) {
  int wave = threadIdx.x >> 5, lane = threadIdx.x & 31;
  long row = (long)blockIdx.x * 8 + wave;
  float s = 0.f;
  for (int d = lane; d < DIMK; d += 32) { float v = (float)z[row * (long)lda + d]; s += v * v; }
#pragma unroll
  for (int off = 16; off > 0; off >>= 1) s += __shfl_xor(s, off, 32);
  if (lane == 0) rn[row] = 0.5f * s;
}

// ksum[b,m] = sum_t kp[b,t,m]   (block b, thread m -> coalesced along m)
__global__ void k_ksum(const __bf16* __restrict__ kp, float* __restrict__ ksum) {
  int b = blockIdx.x, m = threadIdx.x;
  const __bf16* p = kp + ((long)b * TSEQ) * MF + m;
  float s = 0.f;
  for (int t = 0; t < TSEQ; ++t) s += (float)p[(long)t * MF];
  ksum[b * MF + m] = s;
}

// D[row] = dot(qp[row,:], ksum[batch,:])   (one wave32 per row)
__global__ void k_dvec(const __bf16* __restrict__ qp, const float* __restrict__ ksum,
                       float* __restrict__ D) {
  int wave = threadIdx.x >> 5, lane = threadIdx.x & 31;
  long row = (long)blockIdx.x * 8 + wave;
  int b = (int)(row >> 12);   // TSEQ = 4096
  float s = 0.f;
  for (int m = lane; m < MF; m += 32) s += (float)qp[row * MF + m] * ksum[b * MF + m];
#pragma unroll
  for (int off = 16; off > 0; off >>= 1) s += __shfl_xor(s, off, 32);
  if (lane == 0) D[row] = s;
}

// ---------------------------------------------------------------------------
// LDS staging. Tiles are [128 rows][32 k] bf16, row stride 64B.
// This layout lets both WMMA operand fragments be read as 2x b128 ds loads
// matching the ISA VGPR layouts exactly:
//   A frag (16x32): lane L: row = L%16, b128 at row*64 + (L/16)*16 and +32.
//   B frag (32x16): lane L: row = L%16, b128 at row*64 + (L/16)*16 and +32.
// ---------------------------------------------------------------------------
__device__ __forceinline__ void stage_tile_rowmajor(const __bf16* __restrict__ src, int lda,
                                                    long rowBase, int kBase,
                                                    __bf16* __restrict__ lds) {
  int c0 = threadIdx.x * 2;   // 512 chunks of 8 bf16 (16B)
#pragma unroll
  for (int i = 0; i < 2; ++i) {
    int c = c0 + i;
    int r = c >> 2;           // 4 chunks per 32-elem row
    int q = c & 3;
    uint4 val = *(const uint4*)(src + (rowBase + r) * (long)lda + kBase + q * 8);
    *(uint4*)(lds + r * 32 + q * 8) = val;
  }
}

// Transposing stager: source is [k-rows][cols] row-major; LDS gets [col][k].
__device__ __forceinline__ void stage_tile_trans(const __bf16* __restrict__ src, int lda,
                                                 long kRowBase, int colBase,
                                                 __bf16* __restrict__ lds) {
  int c0 = threadIdx.x * 2;
#pragma unroll
  for (int i = 0; i < 2; ++i) {
    int c = c0 + i;
    int t  = c >> 4;          // source k-row 0..31
    int dc = c & 15;          // 8-wide column chunk 0..15
    Chunk8 ch;
    ch.u = *(const uint4*)(src + (kRowBase + t) * (long)lda + colBase + dc * 8);
#pragma unroll
    for (int j = 0; j < 8; ++j)
      lds[(dc * 8 + j) * 32 + t] = ch.e[j];
  }
}

// One 32-deep K step: each wave does a 64x32 C sub-tile = 8 WMMAs.
__device__ __forceinline__ void wmma_tiles(const __bf16* __restrict__ At,
                                           const __bf16* __restrict__ Bt,
                                           v8f acc[4][2], int waveM, int waveN, int lane) {
  int half = lane >> 4;
  int l16  = lane & 15;
  FragBF a[4], b[2];
#pragma unroll
  for (int rt = 0; rt < 4; ++rt) {
    int row = waveM * 64 + rt * 16 + l16;
    const char* base = (const char*)At + row * 64;
    a[rt].u[0] = *(const uint4*)(base + half * 16);
    a[rt].u[1] = *(const uint4*)(base + 32 + half * 16);
  }
#pragma unroll
  for (int ct = 0; ct < 2; ++ct) {
    int col = waveN * 32 + ct * 16 + l16;
    const char* base = (const char*)Bt + col * 64;
    b[ct].u[0] = *(const uint4*)(base + half * 16);
    b[ct].u[1] = *(const uint4*)(base + 32 + half * 16);
  }
#pragma unroll
  for (int rt = 0; rt < 4; ++rt)
#pragma unroll
    for (int ct = 0; ct < 2; ++ct)
      acc[rt][ct] = __builtin_amdgcn_wmma_f32_16x16x32_bf16(
          false, a[rt].v, false, b[ct].v, (short)0, acc[rt][ct], false, false);
}

// ---------------------------------------------------------------------------
// Generic C = A * B^T GEMM (A: MxK row-major, B: NxK row-major), f32 acc,
// fused epilogues. Block tile 128x128, 8 waves (2x4), K step 32,
// double-buffered LDS + prefetch 2 tiles ahead.
// ---------------------------------------------------------------------------
enum { EPI_BIAS_BF16 = 0, EPI_PRM = 1, EPI_SCALED = 2, EPI_BIAS_F32 = 3 };

template <int EPI>
__global__ __launch_bounds__(256)
void k_gemm_wmma(const __bf16* __restrict__ A, const __bf16* __restrict__ Bm,
                 void* __restrict__ Cout,
                 int lda, int ldb, int ldc, int kTiles,
                 const float* __restrict__ bias, const float* __restrict__ rowAux,
                 int rowsPerBatch, long bBatchStride) {
  __shared__ __bf16 At[2][128 * 32];
  __shared__ __bf16 Bt[2][128 * 32];
  int rowBase = blockIdx.y * 128;
  int colBase = blockIdx.x * 128;
  const __bf16* Bp = Bm + (long)(rowBase / rowsPerBatch) * bBatchStride;
  int tid = threadIdx.x, wave = tid >> 5, lane = tid & 31;
  int waveM = wave >> 2, waveN = wave & 3;

  v8f acc[4][2];
  const v8f zero = {0.f, 0.f, 0.f, 0.f, 0.f, 0.f, 0.f, 0.f};
#pragma unroll
  for (int i = 0; i < 4; ++i)
#pragma unroll
    for (int j = 0; j < 2; ++j) acc[i][j] = zero;

  stage_tile_rowmajor(A,  lda, rowBase, 0, At[0]);
  stage_tile_rowmajor(Bp, ldb, colBase, 0, Bt[0]);

  int prow = tid >> 1;   // one prefetch row per thread
  for (int kt = 0; kt < kTiles; ++kt) {
    __syncthreads();                       // buf[kt&1] ready; buf[kt&1 ^ 1] free
    int cur = kt & 1;
    if (kt + 1 < kTiles) {
      stage_tile_rowmajor(A,  lda, rowBase, (kt + 1) * 32, At[cur ^ 1]);
      stage_tile_rowmajor(Bp, ldb, colBase, (kt + 1) * 32, Bt[cur ^ 1]);
    }
    if (kt + 2 < kTiles) {                 // global_prefetch_b8 two tiles ahead
      __builtin_prefetch(A  + (rowBase + prow) * (long)lda + (kt + 2) * 32, 0, 3);
      __builtin_prefetch(Bp + (colBase + prow) * (long)ldb + (kt + 2) * 32, 0, 3);
    }
    wmma_tiles(At[cur], Bt[cur], acc, waveM, waveN, lane);
  }

  int half = lane >> 4, l16 = lane & 15;
#pragma unroll
  for (int rt = 0; rt < 4; ++rt) {
#pragma unroll
    for (int ct = 0; ct < 2; ++ct) {
      int col  = colBase + waveN * 32 + ct * 16 + l16;
      int row0 = rowBase + waveM * 64 + rt * 16 + half * 8;
#pragma unroll
      for (int r = 0; r < 8; ++r) {
        int row = row0 + r;
        float v = acc[rt][ct][r];
        long idx = (long)row * ldc + col;
        if constexpr (EPI == EPI_BIAS_BF16) {
          ((__bf16*)Cout)[idx] = (__bf16)(v + bias[col]);
        } else if constexpr (EPI == EPI_PRM) {
          // exp(w z - |z|^2/2) / sqrt(M), M = 256
          ((__bf16*)Cout)[idx] = (__bf16)(expf(v - rowAux[row]) * 0.0625f);
        } else if constexpr (EPI == EPI_SCALED) {
          ((__bf16*)Cout)[idx] = (__bf16)(v / (rowAux[row] + 1e-8f));
        } else {
          ((float*)Cout)[idx] = v + bias[col];
        }
      }
    }
  }
}

// ---------------------------------------------------------------------------
// kptv[b,d,m] = sum_t v[b,t,d] * kp[b,t,m].  Both operands are K(=t)-major in
// memory, so stage with transpose.  Split-K (8 chunks of 512) + f32 atomics.
// Grid: (EMB/128, MF/128, BATCH*8).
// ---------------------------------------------------------------------------
__global__ __launch_bounds__(256)
void k_kptv_wmma(const __bf16* __restrict__ kqvb, const __bf16* __restrict__ kp,
                 float* __restrict__ outF) {
  __shared__ __bf16 At[2][128 * 32];
  __shared__ __bf16 Bt[2][128 * 32];
  int dBase = blockIdx.x * 128;
  int mBase = blockIdx.y * 128;
  int bb = blockIdx.z >> 3;
  int kc = blockIdx.z & 7;
  long tBase = (long)bb * TSEQ + (long)kc * 512;
  int tid = threadIdx.x, wave = tid >> 5, lane = tid & 31;
  int waveM = wave >> 2, waveN = wave & 3;

  v8f acc[4][2];
  const v8f zero = {0.f, 0.f, 0.f, 0.f, 0.f, 0.f, 0.f, 0.f};
#pragma unroll
  for (int i = 0; i < 4; ++i)
#pragma unroll
    for (int j = 0; j < 2; ++j) acc[i][j] = zero;

  const __bf16* vsrc = kqvb + 2 * EMBD;   // v = cols [1024,1536) of kqv
  stage_tile_trans(vsrc, NKQV, tBase, dBase, At[0]);
  stage_tile_trans(kp,   MF,   tBase, mBase, Bt[0]);

  int pt = tid >> 4;                      // prefetch source-row per thread
  for (int kt = 0; kt < 16; ++kt) {
    __syncthreads();
    int cur = kt & 1;
    if (kt + 1 < 16) {
      stage_tile_trans(vsrc, NKQV, tBase + (kt + 1) * 32, dBase, At[cur ^ 1]);
      stage_tile_trans(kp,   MF,   tBase + (kt + 1) * 32, mBase, Bt[cur ^ 1]);
    }
    if (kt + 2 < 16) {                    // global_prefetch_b8 two tiles ahead
      __builtin_prefetch(vsrc + (tBase + (kt + 2) * 32 + pt) * (long)NKQV + dBase, 0, 3);
      __builtin_prefetch(kp   + (tBase + (kt + 2) * 32 + pt) * (long)MF   + mBase, 0, 3);
    }
    wmma_tiles(At[cur], Bt[cur], acc, waveM, waveN, lane);
  }

  int half = lane >> 4, l16 = lane & 15;
  float* base = outF + (long)bb * EMBD * MF;
#pragma unroll
  for (int rt = 0; rt < 4; ++rt)
#pragma unroll
    for (int ct = 0; ct < 2; ++ct) {
      int col  = mBase + waveN * 32 + ct * 16 + l16;
      int row0 = dBase + waveM * 64 + rt * 16 + half * 8;
#pragma unroll
      for (int r = 0; r < 8; ++r)
        atomicAdd(&base[(long)(row0 + r) * MF + col], acc[rt][ct][r]);
    }
}

// ---------------------------------------------------------------------------
// host side
// ---------------------------------------------------------------------------
extern "C" void kernel_launch(void* const* d_in, const int* in_sizes, int n_in,
                              void* d_out, int out_size, void* d_ws, size_t ws_size,
                              hipStream_t stream) {
  (void)in_sizes; (void)n_in; (void)out_size; (void)ws_size;
  const float* x      = (const float*)d_in[0];
  const float* w_kqv  = (const float*)d_in[1];
  const float* b_kqv  = (const float*)d_in[2];
  const float* w_proj = (const float*)d_in[3];
  const float* b_proj = (const float*)d_in[4];
  const float* w_rf   = (const float*)d_in[5];

  char* ws = (char*)d_ws;
  size_t off = 0;
  auto alloc = [&](size_t bytes) -> char* {
    char* p = ws + off;
    off += (bytes + 255) & ~(size_t)255;
    return p;
  };
  __bf16* xb     = (__bf16*)alloc((size_t)MROWS * DIMK * 2);
  __bf16* wkqvb  = (__bf16*)alloc((size_t)NKQV * DIMK * 2);
  __bf16* wrfb   = (__bf16*)alloc((size_t)MF * DIMK * 2);
  __bf16* wprojb = (__bf16*)alloc((size_t)EMBD * EMBD * 2);
  __bf16* kqvb   = (__bf16*)alloc((size_t)MROWS * NKQV * 2);
  float*  rnk    = (float*) alloc((size_t)MROWS * 4);
  float*  rnq    = (float*) alloc((size_t)MROWS * 4);
  __bf16* kp     = (__bf16*)alloc((size_t)MROWS * MF * 2);
  __bf16* qp     = (__bf16*)alloc((size_t)MROWS * MF * 2);
  float*  ksumA  = (float*) alloc((size_t)BATCH * MF * 4);
  float*  Dv     = (float*) alloc((size_t)MROWS * 4);
  float*  kptvF  = (float*) alloc((size_t)BATCH * EMBD * MF * 4);
  __bf16* kptvB  = (__bf16*)alloc((size_t)BATCH * EMBD * MF * 2);
  __bf16* yb     = (__bf16*)alloc((size_t)MROWS * EMBD * 2);

  // f32 -> bf16 operand conversion
  k_cvt_f32_bf16<<<2048, 256, 0, stream>>>(x,      xb,     MROWS * DIMK);
  k_cvt_f32_bf16<<<512,  256, 0, stream>>>(w_kqv,  wkqvb,  NKQV * DIMK);
  k_cvt_f32_bf16<<<128,  256, 0, stream>>>(w_rf,   wrfb,   MF * DIMK);
  k_cvt_f32_bf16<<<256,  256, 0, stream>>>(w_proj, wprojb, EMBD * EMBD);

  // kqv = x @ w_kqv^T + b   -> bf16 [MROWS x 1536] (cols: k | q | v)
  k_gemm_wmma<EPI_BIAS_BF16><<<dim3(NKQV / 128, MROWS / 128), 256, 0, stream>>>(
      xb, wkqvb, kqvb, DIMK, DIMK, NKQV, DIMK / 32, b_kqv, nullptr, 1 << 30, 0);

  // row norms of k and q
  k_rownorm<<<MROWS / 8, 256, 0, stream>>>(kqvb,        NKQV, rnk);
  k_rownorm<<<MROWS / 8, 256, 0, stream>>>(kqvb + EMBD, NKQV, rnq);

  // kp/qp = exp(z @ w_rf^T - |z|^2/2) / 16
  k_gemm_wmma<EPI_PRM><<<dim3(MF / 128, MROWS / 128), 256, 0, stream>>>(
      kqvb,        wrfb, kp, NKQV, DIMK, MF, DIMK / 32, nullptr, rnk, 1 << 30, 0);
  k_gemm_wmma<EPI_PRM><<<dim3(MF / 128, MROWS / 128), 256, 0, stream>>>(
      kqvb + EMBD, wrfb, qp, NKQV, DIMK, MF, DIMK / 32, nullptr, rnq, 1 << 30, 0);

  // ksum and D
  k_ksum<<<BATCH, MF, 0, stream>>>(kp, ksumA);
  k_dvec<<<MROWS / 8, 256, 0, stream>>>(qp, ksumA, Dv);

  // kptv (split-K, atomic f32), then convert to bf16
  k_zero_f32<<<1024, 256, 0, stream>>>(kptvF, BATCH * EMBD * MF);
  k_kptv_wmma<<<dim3(EMBD / 128, MF / 128, BATCH * 8), 256, 0, stream>>>(kqvb, kp, kptvF);
  k_cvt_f32_bf16<<<1024, 256, 0, stream>>>(kptvF, kptvB, BATCH * EMBD * MF);

  // y = (qp @ kptv^T) / (D + eps)   (B operand batched per 4096 rows)
  k_gemm_wmma<EPI_SCALED><<<dim3(EMBD / 128, MROWS / 128), 256, 0, stream>>>(
      qp, kptvB, yb, MF, MF, EMBD, MF / 32, nullptr, Dv, TSEQ, (long)EMBD * MF);

  // out = y @ w_proj^T + b_proj   (f32 straight into d_out)
  k_gemm_wmma<EPI_BIAS_F32><<<dim3(EMBD / 128, MROWS / 128), 256, 0, stream>>>(
      yb, wprojb, (float*)d_out, EMBD, EMBD, EMBD, EMBD / 32, b_proj, nullptr, 1 << 30, 0);
}
